// MultiHeadSelfAttention_41566693491083
// MI455X (gfx1250) — matrix operand order repaired and checked by reading, not compiled, not run on hardware
//
#include <hip/hip_runtime.h>
#include <hip/hip_bf16.h>

typedef _Float16 h16;
typedef __attribute__((ext_vector_type(8)))  _Float16 v8h;
typedef __attribute__((ext_vector_type(16))) _Float16 v16h;
typedef __attribute__((ext_vector_type(8)))  float    v8f;
typedef __attribute__((ext_vector_type(4)))  unsigned int u32x4;
typedef __attribute__((ext_vector_type(8)))  int          i32x8;
typedef __attribute__((ext_vector_type(4)))  int          i32x4;

#define D_MODEL 1024
#define NUM_H   16
#define D_K     64
#define BATCH   4
#define SEQ     2048
#define M_ROWS  (BATCH * SEQ)   // 8192

static __device__ __forceinline__ v16h mk16(v8h lo, v8h hi) {
  v16h r;
#pragma unroll
  for (int i = 0; i < 8; ++i) { r[i] = lo[i]; r[i + 8] = hi[i]; }
  return r;
}

static __device__ __forceinline__ v8f wmma_f16(v16h a, v16h b, v8f c) {
  return __builtin_amdgcn_wmma_f32_16x16x32_f16(false, a, false, b, (short)0, c,
                                                false, false);
}

// Low 32 bits of a generic LDS pointer == LDS byte offset (ISA 10.2 aperture).
static __device__ __forceinline__ unsigned lds_off(const void* p) {
  return (unsigned)(uintptr_t)p;
}

// TDM 2D tile load: data_size = 8B units.  tile0/tile1 in 8B units,
// stride0 (dim0 line stride) in 8B units.  D# per CDNA5 ISA ch.8.
// amdgpu-toolchain (clang-23) 6-arg builtin form.
static __device__ __forceinline__ void tdm_load_2d(unsigned lds_addr,
                                                   const void* gptr,
                                                   unsigned tile0, unsigned tile1,
                                                   unsigned long long stride0) {
  unsigned long long ga = (unsigned long long)(uintptr_t)gptr;
  const unsigned td0 = 1u << 20, td1 = 1u << 20;  // generous OOB bounds
  u32x4 g0;
  g0[0] = 1u;                                   // count=1, user descriptor
  g0[1] = lds_addr;                             // [63:32] lds_addr
  g0[2] = (unsigned)ga;                         // [95:64] global_addr lo
  g0[3] = (unsigned)((ga >> 32) & 0x01FFFFFFu)  // [120:96] global_addr hi
          | (2u << 30);                         // [127:126] type=2 (image)
  i32x8 g1;
  g1[0] = (int)(3u << 16);                      // data_size=3 (8B); mask=0
  g1[1] = (int)((td0 & 0xFFFFu) << 16);         // [63:48] tensor_dim0 lo16
  g1[2] = (int)((td0 >> 16) | ((td1 & 0xFFFFu) << 16));
  g1[3] = (int)((td1 >> 16) | (tile0 << 16));   // [127:112] tile_dim0
  g1[4] = (int)(tile1 & 0xFFFFu);               // [143:128] tile_dim1 (tile2=0)
  g1[5] = (int)(unsigned)(stride0 & 0xFFFFFFFFull);
  g1[6] = (int)(unsigned)((stride0 >> 32) & 0xFFFFull);
  g1[7] = 0;
  i32x4 g2 = {0, 0, 0, 0};
  i32x4 g3 = {0, 0, 0, 0};
  i32x8 g4 = {0, 0, 0, 0, 0, 0, 0, 0};
  __builtin_amdgcn_tensor_load_to_lds(g0, g1, g2, g3, g4, 0);
}

// ---------------------------------------------------------------- converts
__global__ void cvt_f32_f16(const float* __restrict__ src, h16* __restrict__ dst,
                            int n) {
  int i = blockIdx.x * 256 + threadIdx.x;
  if (i < n) dst[i] = (h16)src[i];
}

__global__ void wtrans_f16(const float* __restrict__ W, h16* __restrict__ WT) {
  __shared__ float tile[16][17];
  int bx = blockIdx.x * 16, by = blockIdx.y * 16;
  int tx = threadIdx.x, ty = threadIdx.y;
  tile[ty][tx] = W[(size_t)(by + ty) * D_MODEL + bx + tx];
  __syncthreads();
  WT[(size_t)(bx + ty) * D_MODEL + by + tx] = (h16)tile[tx][ty];
}

// ---------------------------------------------------------------- QKV GEMM
// Block tile 128x64, 4 waves, wave tile 32x64 (each B fragment feeds 2 WMMAs).
// vmode 0: Out[((b*16+h)*2048+t)*64+d]   vmode 1: Out[((b*16+h)*64+d)*2048+t]
__global__ __launch_bounds__(128) void qkv_gemm(
    const h16* __restrict__ Xh, const h16* __restrict__ WT,
    const float* __restrict__ bias, h16* __restrict__ Out, float scale,
    int vmode) {
  __shared__ h16 At[128 * 32];
  __shared__ h16 Bt[64 * 32];
  const int tid = threadIdx.x;
  const int wave = tid >> 5, lane = tid & 31;
  const int r0 = blockIdx.x * 128, n0 = blockIdx.y * 64;
  const int colh = lane & 15, hs = lane >> 4;

  v8f acc[2][4];
#pragma unroll
  for (int m = 0; m < 2; ++m)
#pragma unroll
    for (int t = 0; t < 4; ++t)
#pragma unroll
      for (int r = 0; r < 8; ++r) acc[m][t][r] = 0.f;

  for (int kk = 0; kk < D_MODEL; kk += 32) {
    {  // At: 128 rows x 32 halves, one row per thread
      const h16* xs = Xh + (size_t)(r0 + tid) * D_MODEL + kk;
#pragma unroll
      for (int i = 0; i < 4; ++i)
        *(v8h*)(At + tid * 32 + i * 8) = *(const v8h*)(xs + i * 8);
      // Bt: 64 rows x 32 halves, half row per thread
      const h16* wsrc = WT + (size_t)(n0 + (tid >> 1)) * D_MODEL + kk +
                        (tid & 1) * 16;
      *(v8h*)(Bt + (tid >> 1) * 32 + (tid & 1) * 16) = *(const v8h*)(wsrc);
      *(v8h*)(Bt + (tid >> 1) * 32 + (tid & 1) * 16 + 8) =
          *(const v8h*)(wsrc + 8);
    }
    __syncthreads();

    const int kh = hs * 8, kb = hs * 16;
    v16h af[2];
#pragma unroll
    for (int m = 0; m < 2; ++m) {
      const h16* arow = At + (wave * 32 + m * 16 + colh) * 32;
      af[m] = mk16(*(const v8h*)(arow + kb), *(const v8h*)(arow + kb + 8));
    }
#pragma unroll
    for (int nt = 0; nt < 4; ++nt) {
      const h16* brow = Bt + (nt * 16 + colh) * 32;
      v16h bf = mk16(*(const v8h*)(brow + kb), *(const v8h*)(brow + kb + 8));
#pragma unroll
      for (int m = 0; m < 2; ++m) acc[m][nt] = wmma_f16(af[m], bf, acc[m][nt]);
    }
    __syncthreads();
  }

#pragma unroll
  for (int nt = 0; nt < 4; ++nt) {
    const int n = n0 + nt * 16 + colh;
    const float bv = bias[n];
    const int hh = n >> 6, d = n & 63;
#pragma unroll
    for (int m = 0; m < 2; ++m)
#pragma unroll
      for (int r = 0; r < 8; ++r) {
        const int row = r0 + wave * 32 + m * 16 + 8 * hs + r;
        const int bb = row >> 11, t = row & (SEQ - 1);
        const float val = (acc[m][nt][r] + bv) * scale;
        size_t idx = vmode ? ((size_t)((bb * NUM_H + hh) * D_K + d)) * SEQ + t
                           : ((size_t)((bb * NUM_H + hh) * SEQ + t)) * D_K + d;
        Out[idx] = (h16)val;
      }
  }
}

// ---------------------------------------------------------------- attention
// grid (T/64, B*H), 128 threads = 4 waves; wave -> 16-query tile.
// K/V staged to LDS by the Tensor Data Mover, double-buffered.
__global__ __launch_bounds__(128) void attn_kernel(
    const h16* __restrict__ Qf, const h16* __restrict__ Kf,
    const h16* __restrict__ VTf, const int* __restrict__ mask,
    h16* __restrict__ Ctx) {
  __shared__ h16 Kt[2][64 * 64];   // [key][dk]
  __shared__ h16 Vt[2][64 * 64];   // [dk][key]
  __shared__ float biasS[2][64];
  __shared__ h16 Pb[4][16 * 32];   // per-wave P tile [qrow][key-in-group]

  const int tid = threadIdx.x;
  const int wave = tid >> 5, lane = tid & 31;
  const int bh = blockIdx.y;
  const int b = bh >> 4, h = bh & 15;
  const int q0 = blockIdx.x * 64 + wave * 16;
  const int colh = lane & 15, hs = lane >> 4;

  const h16* Qb = Qf + (size_t)bh * SEQ * D_K;
  const h16* Kb = Kf + (size_t)bh * SEQ * D_K;
  const h16* Vb = VTf + (size_t)bh * D_K * SEQ;

  v16h qf0, qf1;
  {
    const h16* qrow = Qb + (size_t)(q0 + colh) * D_K;
    const int kb = hs * 16;
    qf0 = mk16(*(const v8h*)(qrow + kb), *(const v8h*)(qrow + kb + 8));
    qf1 = mk16(*(const v8h*)(qrow + 32 + kb), *(const v8h*)(qrow + 32 + kb + 8));
  }

  float m_i[8], l_i[8];
  v8f acc[4];
#pragma unroll
  for (int r = 0; r < 8; ++r) { m_i[r] = -1e30f; l_i[r] = 0.f; }
#pragma unroll
  for (int t = 0; t < 4; ++t)
#pragma unroll
    for (int r = 0; r < 8; ++r) acc[t][r] = 0.f;

  // prologue: stage chunk 0 into buffer 0
  if (wave == 0) {
    // K chunk: contiguous 8KB -> 1 line of 1024 x 8B
    tdm_load_2d(lds_off(&Kt[0][0]), Kb, 1024, 1, 1024);
    // V^T chunk: 64 rows x 128B, row stride T*2B = 512 x 8B
    tdm_load_2d(lds_off(&Vt[0][0]), Vb, 16, 64, SEQ / 4);
  }
  if (tid < 64)
    biasS[0][tid] = (mask[b * SEQ + tid] != 0) ? 0.f : -1e9f;

  for (int kc = 0; kc < SEQ / 64; ++kc) {
    const int buf = kc & 1;
    if (wave == 0) __builtin_amdgcn_s_wait_tensorcnt(0);
    __syncthreads();  // buffer `buf` (TDM) + biasS[buf] now visible to all

    if (kc + 1 < SEQ / 64) {  // prefetch next chunk into the other buffer
      if (wave == 0) {
        const h16* knext = Kb + (size_t)(kc + 1) * 64 * D_K;
        const h16* vnext = Vb + (size_t)(kc + 1) * 64;
        tdm_load_2d(lds_off(&Kt[buf ^ 1][0]), knext, 1024, 1, 1024);
        tdm_load_2d(lds_off(&Vt[buf ^ 1][0]), vnext, 16, 64, SEQ / 4);
      }
      if (tid < 64)
        biasS[buf ^ 1][tid] =
            (mask[b * SEQ + (kc + 1) * 64 + tid] != 0) ? 0.f : -1e9f;
    }

    const h16* KtB = Kt[buf];
    const h16* VtB = Vt[buf];

#pragma unroll
    for (int g = 0; g < 2; ++g) {  // 32-key groups
#pragma unroll
      for (int sub = 0; sub < 2; ++sub) {  // 16-key score tiles
        const int kt = g * 32 + sub * 16;
        const int kb = hs * 16;
        const h16* krow = KtB + (kt + colh) * 64;
        v16h bf0 = mk16(*(const v8h*)(krow + kb), *(const v8h*)(krow + kb + 8));
        v16h bf1 = mk16(*(const v8h*)(krow + 32 + kb),
                        *(const v8h*)(krow + 32 + kb + 8));
        v8f zero;
#pragma unroll
        for (int r = 0; r < 8; ++r) zero[r] = 0.f;
        v8f s = wmma_f16(qf0, bf0, zero);
        s = wmma_f16(qf1, bf1, s);
        const float kbias = biasS[buf][kt + colh];
#pragma unroll
        for (int r = 0; r < 8; ++r) {
          float sv = s[r] + kbias;
          float tmx = sv;
          tmx = fmaxf(tmx, __shfl_xor(tmx, 1, 32));
          tmx = fmaxf(tmx, __shfl_xor(tmx, 2, 32));
          tmx = fmaxf(tmx, __shfl_xor(tmx, 4, 32));
          tmx = fmaxf(tmx, __shfl_xor(tmx, 8, 32));
          const float nm = fmaxf(m_i[r], tmx);
          const float alpha = __expf(m_i[r] - nm);
          m_i[r] = nm;
          const float pv = __expf(sv - nm);
          float rs = pv;
          rs += __shfl_xor(rs, 1, 32);
          rs += __shfl_xor(rs, 2, 32);
          rs += __shfl_xor(rs, 4, 32);
          rs += __shfl_xor(rs, 8, 32);
          l_i[r] = l_i[r] * alpha + rs;
#pragma unroll
          for (int t4 = 0; t4 < 4; ++t4) acc[t4][r] *= alpha;
          Pb[wave][(r + 8 * hs) * 32 + sub * 16 + colh] = (h16)pv;
        }
      }
      {  // P(16x32) x V(32x64)
        const int kh = hs * 8, kb = hs * 16;
        const h16* prow = &Pb[wave][colh * 32];
        v16h pf = mk16(*(const v8h*)(prow + kb), *(const v8h*)(prow + kb + 8));
#pragma unroll
        for (int dt = 0; dt < 4; ++dt) {
          const h16* vrow = VtB + (dt * 16 + colh) * 64 + g * 32;
          v16h bf = mk16(*(const v8h*)(vrow + kb), *(const v8h*)(vrow + kb + 8));
          acc[dt] = wmma_f16(pf, bf, acc[dt]);
        }
      }
    }
    __syncthreads();  // everyone done with buffer `buf` before it is refilled
  }

#pragma unroll
  for (int dt = 0; dt < 4; ++dt) {
#pragma unroll
    for (int r = 0; r < 8; ++r) {
      const float o = acc[dt][r] * (1.0f / l_i[r]);
      const int row = q0 + 8 * hs + r;
      const size_t idx =
          ((size_t)(b * SEQ + row)) * D_MODEL + h * D_K + dt * 16 + colh;
      Ctx[idx] = (h16)o;
    }
  }
}

// ---------------------------------------------------------------- out proj
__global__ __launch_bounds__(128) void out_gemm(
    const h16* __restrict__ Ah, const h16* __restrict__ WT,
    const float* __restrict__ bias, float* __restrict__ Out) {
  __shared__ h16 At[128 * 32];
  __shared__ h16 Bt[64 * 32];
  const int tid = threadIdx.x;
  const int wave = tid >> 5, lane = tid & 31;
  const int r0 = blockIdx.x * 128, n0 = blockIdx.y * 64;
  const int colh = lane & 15, hs = lane >> 4;

  v8f acc[2][4];
#pragma unroll
  for (int m = 0; m < 2; ++m)
#pragma unroll
    for (int t = 0; t < 4; ++t)
#pragma unroll
      for (int r = 0; r < 8; ++r) acc[m][t][r] = 0.f;

  for (int kk = 0; kk < D_MODEL; kk += 32) {
    {
      const h16* xs = Ah + (size_t)(r0 + tid) * D_MODEL + kk;
#pragma unroll
      for (int i = 0; i < 4; ++i)
        *(v8h*)(At + tid * 32 + i * 8) = *(const v8h*)(xs + i * 8);
      const h16* wsrc = WT + (size_t)(n0 + (tid >> 1)) * D_MODEL + kk +
                        (tid & 1) * 16;
      *(v8h*)(Bt + (tid >> 1) * 32 + (tid & 1) * 16) = *(const v8h*)(wsrc);
      *(v8h*)(Bt + (tid >> 1) * 32 + (tid & 1) * 16 + 8) =
          *(const v8h*)(wsrc + 8);
    }
    __syncthreads();

    const int kh = hs * 8, kb = hs * 16;
    v16h af[2];
#pragma unroll
    for (int m = 0; m < 2; ++m) {
      const h16* arow = At + (wave * 32 + m * 16 + colh) * 32;
      af[m] = mk16(*(const v8h*)(arow + kb), *(const v8h*)(arow + kb + 8));
    }
#pragma unroll
    for (int nt = 0; nt < 4; ++nt) {
      const h16* brow = Bt + (nt * 16 + colh) * 32;
      v16h bf = mk16(*(const v8h*)(brow + kb), *(const v8h*)(brow + kb + 8));
#pragma unroll
      for (int m = 0; m < 2; ++m) acc[m][nt] = wmma_f16(af[m], bf, acc[m][nt]);
    }
    __syncthreads();
  }

#pragma unroll
  for (int nt = 0; nt < 4; ++nt) {
    const int n = n0 + nt * 16 + colh;
    const float bv = bias[n];
#pragma unroll
    for (int m = 0; m < 2; ++m)
#pragma unroll
      for (int r = 0; r < 8; ++r) {
        const int row = r0 + wave * 32 + m * 16 + 8 * hs + r;
        Out[(size_t)row * D_MODEL + n] = acc[m][nt][r] + bv;
      }
  }
}

// ---------------------------------------------------------------- launcher
extern "C" void kernel_launch(void* const* d_in, const int* in_sizes, int n_in,
                              void* d_out, int out_size, void* d_ws,
                              size_t ws_size, hipStream_t stream) {
  const float* x = (const float*)d_in[0];
  const int* mask = (const int*)d_in[1];
  const float* wq = (const float*)d_in[2];
  const float* bq = (const float*)d_in[3];
  const float* wk = (const float*)d_in[4];
  const float* bk = (const float*)d_in[5];
  const float* wv = (const float*)d_in[6];
  const float* bv = (const float*)d_in[7];
  const float* wo = (const float*)d_in[8];
  const float* bo = (const float*)d_in[9];

  char* ws = (char*)d_ws;
  const size_t XH_B = (size_t)M_ROWS * D_MODEL * 2;   // 16 MB
  const size_t WT_B = (size_t)D_MODEL * D_MODEL * 2;  // 2 MB
  h16* Xh  = (h16*)ws; ws += XH_B;
  h16* WTq = (h16*)ws; ws += WT_B;
  h16* WTk = (h16*)ws; ws += WT_B;
  h16* WTv = (h16*)ws; ws += WT_B;
  h16* WTo = (h16*)ws; ws += WT_B;
  h16* Qf  = (h16*)ws; ws += XH_B;  // [B,H,T,64]
  h16* Kf  = (h16*)ws; ws += XH_B;  // [B,H,T,64]
  h16* VTf = (h16*)ws; ws += XH_B;  // [B,H,64,T]
  h16* Ctx = (h16*)ws; ws += XH_B;  // [B*T,1024]

  {
    const int n = M_ROWS * D_MODEL;
    cvt_f32_f16<<<(n + 255) / 256, 256, 0, stream>>>(x, Xh, n);
    dim3 g(64, 64), bdim(16, 16);
    wtrans_f16<<<g, bdim, 0, stream>>>(wq, WTq);
    wtrans_f16<<<g, bdim, 0, stream>>>(wk, WTk);
    wtrans_f16<<<g, bdim, 0, stream>>>(wv, WTv);
    wtrans_f16<<<g, bdim, 0, stream>>>(wo, WTo);
  }
  {
    dim3 g(M_ROWS / 128, D_MODEL / 64);
    qkv_gemm<<<g, 128, 0, stream>>>(Xh, WTq, bq, Qf, 0.125f, 0);
    qkv_gemm<<<g, 128, 0, stream>>>(Xh, WTk, bk, Kf, 1.0f, 0);
    qkv_gemm<<<g, 128, 0, stream>>>(Xh, WTv, bv, VTf, 1.0f, 1);
  }
  {
    dim3 g(SEQ / 64, BATCH * NUM_H);
    attn_kernel<<<g, 128, 0, stream>>>(Qf, Kf, VTf, mask, Ctx);
  }
  {
    dim3 g(M_ROWS / 128, D_MODEL / 64);
    out_gemm<<<g, 128, 0, stream>>>(Ctx, WTo, bo, (float*)d_out);
  }
}
